// Seq2Seq_40819369181605
// MI455X (gfx1250) — matrix operand order repaired and checked by reading, not compiled, not run on hardware
//
#include <hip/hip_runtime.h>
#include <hip/hip_bf16.h>

// ---------------- problem constants (from reference) ----------------
#define Bb   64
#define Tt   800
#define Ee   1024
#define Dd   1024
#define DEe  256
#define Aa   256
#define Vv   8000
#define Ll   150
#define SCALE_F 2.0f
#define KCAT (DEe + Ee + Dd)   // 2304  (x_t | ctx | h)
#define KHC  (Dd + Ee)         // 2048  (h | ctx)

// ---------------- WMMA types ----------------
typedef __attribute__((ext_vector_type(16))) __bf16 v16bf;
typedef __attribute__((ext_vector_type(8)))  float  v8f;

union FragBF { v16bf v; unsigned int u[8]; };

#if defined(__has_builtin)
#if __has_builtin(__builtin_amdgcn_sched_barrier)
#define SCHED_FENCE() __builtin_amdgcn_sched_barrier(0)
#endif
#endif
#ifndef SCHED_FENCE
#define SCHED_FENCE()
#endif

__device__ __forceinline__ unsigned short f2bf(float x) {
  unsigned int u = __float_as_uint(x);
  unsigned int r = u + 0x7FFFu + ((u >> 16) & 1u);   // round-to-nearest-even
  return (unsigned short)(r >> 16);
}
__device__ __forceinline__ float bf2f(unsigned short h) {
  return __uint_as_float(((unsigned int)h) << 16);
}
__device__ __forceinline__ float sigmoidf_(float x) {
  return 1.0f / (1.0f + __expf(-x));
}

// =====================================================================
// Unified bf16 WMMA GEMM:  C[M,N] = act( A[M,K] * W[N,K]^T + bias )
// Register-blocked (1 m-tile x 4 n-tiles per wave), software-pipelined
// with double-buffered fragments + sched_barriers. LDW is a template
// parameter so the 4 B-tile streams are one pointer + compile-time
// offsets (j*16*LDW elems, folded into GLOBAL_LOAD's 24-bit IOFFSET):
// only TWO live pointers in the loop -> no 64-bit moves, no spills.
// =====================================================================
#define NT 4   // n-tiles per wave

__device__ __forceinline__ void load_a_frag(FragBF& a,
                                            const unsigned short* ap,
                                            int aHalf)
{
#pragma unroll
  for (int g = 0; g < 8; ++g) {
    // A 16x32 layout (matches B): VGPR g holds K=half*16+g*2, +1
    const int ka = (aHalf << 1) + (g << 1);
    a.u[g] = *(const unsigned int*)(ap + ka);
  }
}

template <int LDW>
__device__ __forceinline__ void load_b_frags(FragBF* b,
                                             const unsigned short* bp)
{
  // B 32x16 k order used here (contiguous, same as A): lanes0-15 K=0..15, lanes16-31 K=16..31; 2/dword
#pragma unroll
  for (int j = 0; j < NT; ++j)
#pragma unroll
    for (int g = 0; g < 8; ++g)
      b[j].u[g] = *(const unsigned int*)(bp + j * 16 * LDW + (g << 1));
}

template <int LDW>
__global__ void wmma_gemm_bf16(const unsigned short* __restrict__ A, int lda,
                               const unsigned short* __restrict__ W,
                               const float* __restrict__ bias,
                               float* __restrict__ Cf, long ldc,
                               unsigned short* __restrict__ Cbf, int ldcb,
                               int M, int N, int K, int doTanh)
{
  const int wave = blockIdx.x * (blockDim.x >> 5) + (threadIdx.x >> 5);
  const int lane = threadIdx.x & 31;
  const int tilesM = M >> 4;
  const int tasks  = tilesM * (N >> 6);      // N blocked by 64
  if (wave >= tasks) return;                 // wave-uniform: EXEC stays all-1s

  const int tm  = wave % tilesM;
  const int tn4 = wave / tilesM;
  const int m0  = tm << 4;
  const int n0  = tn4 << 6;                  // first of 4 consecutive 16-col tiles

  const int mrow  = m0 + (lane & 15);
  const int nlane = lane & 15;
  const int aHalf = (lane >> 4) << 3;        // 0 or 8   (A-frag K sub-offset)
  const int bOff  = (lane >> 4) << 4;        // 0 or 16  (B-frag K sub-offset)

  v8f acc[NT];
#pragma unroll
  for (int j = 0; j < NT; ++j) acc[j] = (v8f){};

  // ---- two live stream pointers; B tile offsets are compile-time ----
  const unsigned short* ap = A + (size_t)mrow * lda;
  const unsigned short* bp = W + (size_t)(n0 + nlane) * LDW + bOff;

  // ---- software pipeline: prologue loads k-step 0 ----
  FragBF aC, bC[NT];
  load_a_frag(aC, ap, aHalf);
  load_b_frags<LDW>(bC, bp);
  ap += 32; bp += 32;

#pragma unroll 2
  for (int k0 = 32; k0 < K; k0 += 32) {
    FragBF aN, bN[NT];
    load_a_frag(aN, ap, aHalf);             // prefetch next k-step
    load_b_frags<LDW>(bN, bp);
    ap += 32; bp += 32;
    SCHED_FENCE();                          // keep prefetch loads above math
#pragma unroll
    for (int j = 0; j < NT; ++j)
      acc[j] = __builtin_amdgcn_wmma_f32_16x16x32_bf16(false, aC.v, false,
                                                       bC[j].v, (short)0,
                                                       acc[j], false, false);
    SCHED_FENCE();
    aC = aN;
#pragma unroll
    for (int j = 0; j < NT; ++j) bC[j] = bN[j];
  }
  // ---- epilogue: last k-step ----
#pragma unroll
  for (int j = 0; j < NT; ++j)
    acc[j] = __builtin_amdgcn_wmma_f32_16x16x32_bf16(false, aC.v, false,
                                                     bC[j].v, (short)0,
                                                     acc[j], false, false);

#pragma unroll
  for (int j = 0; j < NT; ++j) {
    const int n  = n0 + (j << 4) + nlane;
    const float bv = bias ? bias[n] : 0.0f;
#pragma unroll
    for (int r = 0; r < 8; ++r) {
      const int m = m0 + r + ((lane >> 4) << 3);
      float v = acc[j][r] + bv;
      if (doTanh) v = tanhf(v);
      if (Cf)  Cf[(size_t)m * ldc + n]   = v;
      if (Cbf) Cbf[(size_t)m * ldcb + n] = f2bf(v);
    }
  }
}

// =====================================================================
// One-time conversion / init kernels
// =====================================================================
__global__ void conv_f32_to_bf16(const float* __restrict__ src,
                                 unsigned short* __restrict__ dst, long n)
{
  long i = (long)blockIdx.x * blockDim.x + threadIdx.x;
  if (i < n) dst[i] = f2bf(src[i]);
}

// Wcat[n][k] : k<1280 -> W_ih[n][k] ; else W_hh[n][k-1280]
__global__ void conv_wcat(const float* __restrict__ wih,
                          const float* __restrict__ whh,
                          unsigned short* __restrict__ wcat)
{
  long i = (long)blockIdx.x * blockDim.x + threadIdx.x;
  if (i >= (long)(4 * Dd) * KCAT) return;
  int n = (int)(i / KCAT), k = (int)(i % KCAT);
  float v = (k < (DEe + Ee)) ? wih[(size_t)n * (DEe + Ee) + k]
                             : whh[(size_t)n * Dd + (k - (DEe + Ee))];
  wcat[i] = f2bf(v);
}

// dst[cols][rows] = bf16( src[rows][cols] )
__global__ void conv_transpose_bf16(const float* __restrict__ src,
                                    unsigned short* __restrict__ dst,
                                    int rows, int cols)
{
  long i = (long)blockIdx.x * blockDim.x + threadIdx.x;
  if (i >= (long)rows * cols) return;
  int r = (int)(i / cols), c = (int)(i % cols);
  dst[(size_t)c * rows + r] = f2bf(src[i]);
}

__global__ void init_state(float* __restrict__ h, float* __restrict__ c,
                           float* __restrict__ ctx, float* __restrict__ bsum,
                           const float* __restrict__ b_ih,
                           const float* __restrict__ b_hh)
{
  int i = blockIdx.x * blockDim.x + threadIdx.x;
  if (i < Bb * Dd) { h[i] = 0.0f; c[i] = 0.0f; }
  if (i < Bb * Ee) ctx[i] = 0.0f;
  if (i < 4 * Dd)  bsum[i] = b_ih[i] + b_hh[i];
}

// =====================================================================
// Per-step kernels
// =====================================================================
// Build concatenated LSTM input [x_t | ctx | h] in bf16.
__global__ void prep_step(const float* __restrict__ emb,
                          const int* __restrict__ tokens, int t,
                          const float* __restrict__ h,
                          const float* __restrict__ ctx,
                          unsigned short* __restrict__ concatA)
{
  int idx = blockIdx.x * blockDim.x + threadIdx.x;
  if (idx >= Bb * KCAT) return;
  int b = idx / KCAT, c = idx % KCAT;
  float v;
  if (c < DEe) {
    int tok = tokens[b * Ll + t];
    v = emb[(size_t)tok * DEe + c];
  } else if (c < DEe + Ee) {
    v = ctx[b * Ee + (c - DEe)];
  } else {
    v = h[b * Dd + (c - DEe - Ee)];
  }
  concatA[idx] = f2bf(v);
}

// LSTM cell elementwise (gates order i,f,g,o); writes h (fp32 + bf16-to-hc), c.
__global__ void lstm_update(const float* __restrict__ gates,
                            float* __restrict__ h, float* __restrict__ c,
                            unsigned short* __restrict__ hc)
{
  int idx = blockIdx.x * blockDim.x + threadIdx.x;
  if (idx >= Bb * Dd) return;
  int b = idx / Dd, d = idx % Dd;
  const float* g = gates + (size_t)b * (4 * Dd);
  float ig = sigmoidf_(g[d]);
  float fg = sigmoidf_(g[Dd + d]);
  float gg = tanhf(g[2 * Dd + d]);
  float og = sigmoidf_(g[3 * Dd + d]);
  float cn = fg * c[idx] + ig * gg;
  float hn = og * tanhf(cn);
  c[idx] = cn;
  h[idx] = hn;
  hc[(size_t)b * KHC + d] = f2bf(hn);     // [h | ctx] buffer, h part
}

// energy = comp_enc . q ; masked scaled softmax over T -> attn (fp32)
__global__ void attn_softmax(const unsigned short* __restrict__ comp,
                             const unsigned short* __restrict__ qbf,
                             const int* __restrict__ state_len,
                             float* __restrict__ attn)
{
  __shared__ float sQ[Aa];
  __shared__ float sE[Tt];
  __shared__ float red[256];
  const int b = blockIdx.x, tid = threadIdx.x;

  if (tid < Aa) sQ[tid] = bf2f(qbf[b * Aa + tid]);
  __syncthreads();

  const int sl = state_len[b];
  for (int t = tid; t < Tt; t += 256) {
    float e;
    if (t >= sl) {
      e = -INFINITY;
    } else {
      const unsigned int* row =
          (const unsigned int*)(comp + ((size_t)b * Tt + t) * Aa);
      float acc = 0.0f;
#pragma unroll 4
      for (int a = 0; a < Aa / 2; ++a) {
        unsigned int u = row[a];
        acc += bf2f((unsigned short)u)         * sQ[2 * a];
        acc += bf2f((unsigned short)(u >> 16)) * sQ[2 * a + 1];
      }
      e = acc * SCALE_F;
    }
    sE[t] = e;
  }
  __syncthreads();

  float mx = -INFINITY;
  for (int t = tid; t < Tt; t += 256) mx = fmaxf(mx, sE[t]);
  red[tid] = mx; __syncthreads();
  for (int s = 128; s > 0; s >>= 1) {
    if (tid < s) red[tid] = fmaxf(red[tid], red[tid + s]);
    __syncthreads();
  }
  mx = red[0];
  __syncthreads();

  float sum = 0.0f;
  for (int t = tid; t < Tt; t += 256) {
    float p = __expf(sE[t] - mx);            // exp(-inf) == 0 handles mask
    sE[t] = p;
    sum += p;
  }
  red[tid] = sum; __syncthreads();
  for (int s = 128; s > 0; s >>= 1) {
    if (tid < s) red[tid] += red[tid + s];
    __syncthreads();
  }
  const float inv = 1.0f / red[0];
  for (int t = tid; t < Tt; t += 256) attn[b * Tt + t] = sE[t] * inv;
}

// ctx[b,e] = sum_t attn[b,t] * listener_bf16[b,t,e]   (L2-resident bf16 stream)
__global__ void ctx_kernel(const float* __restrict__ attn,
                           const unsigned short* __restrict__ lis_bf,
                           float* __restrict__ ctx,
                           unsigned short* __restrict__ hc)
{
  __shared__ float sA[Tt];
  const int b = blockIdx.x >> 2;
  const int e = ((blockIdx.x & 3) << 8) + threadIdx.x;
  for (int t = threadIdx.x; t < Tt; t += 256) sA[t] = attn[b * Tt + t];
  __syncthreads();

  const unsigned short* base = lis_bf + (size_t)b * Tt * Ee + e;
  float acc = 0.0f;
  for (int t = 0; t < Tt; ++t)
    acc += sA[t] * bf2f(base[(size_t)t * Ee]);

  ctx[b * Ee + e] = acc;
  hc[(size_t)b * KHC + Dd + e] = f2bf(acc); // [h | ctx] buffer, ctx part
}

// =====================================================================
// Host launcher
// =====================================================================
static inline size_t align_up(size_t x) { return (x + 255) & ~(size_t)255; }

extern "C" void kernel_launch(void* const* d_in, const int* in_sizes, int n_in,
                              void* d_out, int out_size, void* d_ws, size_t ws_size,
                              hipStream_t stream)
{
  (void)in_sizes; (void)n_in; (void)out_size; (void)ws_size;

  const float* listener = (const float*)d_in[0];   // [B,T,E]
  const int*   state_len= (const int*)  d_in[1];   // [B]
  const int*   tokens   = (const int*)  d_in[2];   // [B,L]
  const float* embedding= (const float*)d_in[3];   // [V,DE]
  const float* W_ih     = (const float*)d_in[4];   // [4D, DE+E]
  const float* W_hh     = (const float*)d_in[5];   // [4D, D]
  const float* b_ih     = (const float*)d_in[6];
  const float* b_hh     = (const float*)d_in[7];
  const float* phi      = (const float*)d_in[8];   // [D, A]
  const float* psi      = (const float*)d_in[9];   // [E, A]
  const float* b_psi    = (const float*)d_in[10];
  const float* W_cls    = (const float*)d_in[11];  // [D+E, V]
  const float* b_cls    = (const float*)d_in[12];
  float* out = (float*)d_out;                      // [B, L, V]

  // ---- workspace bump allocator ----
  char* p = (char*)d_ws;
  auto alloc = [&](size_t bytes) -> char* {
    char* r = p; p += align_up(bytes); return r;
  };
  unsigned short* lis_bf  = (unsigned short*)alloc((size_t)Bb*Tt*Ee * 2);  // 105 MB, L2-resident
  unsigned short* comp_bf = (unsigned short*)alloc((size_t)Bb*Tt*Aa * 2);  //  26 MB
  unsigned short* wcat_bf = (unsigned short*)alloc((size_t)4*Dd*KCAT * 2); //  19 MB
  unsigned short* wcls_bf = (unsigned short*)alloc((size_t)Vv*KHC   * 2);  //  33 MB
  unsigned short* phiT_bf = (unsigned short*)alloc((size_t)Aa*Dd * 2);
  unsigned short* psiT_bf = (unsigned short*)alloc((size_t)Aa*Ee * 2);
  float*          bsum    = (float*)        alloc((size_t)4*Dd * 4);
  unsigned short* concatA = (unsigned short*)alloc((size_t)Bb*KCAT * 2);
  unsigned short* hc_bf   = (unsigned short*)alloc((size_t)Bb*KHC * 2);
  float*          gates   = (float*)        alloc((size_t)Bb*4*Dd * 4);
  float*          hbuf    = (float*)        alloc((size_t)Bb*Dd * 4);
  float*          cbuf    = (float*)        alloc((size_t)Bb*Dd * 4);
  float*          ctxbuf  = (float*)        alloc((size_t)Bb*Ee * 4);
  unsigned short* q_bf    = (unsigned short*)alloc((size_t)Bb*Aa * 2);
  float*          attn    = (float*)        alloc((size_t)Bb*Tt * 4);

  const int TPB = 256;
  auto nb = [&](long n) { return (unsigned)((n + TPB - 1) / TPB); };
  // GEMM task count: (M/16) * (N/64) waves, 8 waves per 256-thread block
  auto gemm_blocks = [](int M, int N) {
    int tasks = (M >> 4) * (N >> 6);
    return (unsigned)((tasks + 7) / 8);
  };

  // ---- one-time: init + weight/feature conversion ----
  init_state<<<nb(Bb*Dd), TPB, 0, stream>>>(hbuf, cbuf, ctxbuf, bsum, b_ih, b_hh);
  conv_f32_to_bf16<<<nb((long)Bb*Tt*Ee), TPB, 0, stream>>>(listener, lis_bf, (long)Bb*Tt*Ee);
  conv_wcat<<<nb((long)4*Dd*KCAT), TPB, 0, stream>>>(W_ih, W_hh, wcat_bf);
  conv_transpose_bf16<<<nb((long)KHC*Vv), TPB, 0, stream>>>(W_cls, wcls_bf, KHC, Vv);
  conv_transpose_bf16<<<nb((long)Dd*Aa), TPB, 0, stream>>>(phi, phiT_bf, Dd, Aa);
  conv_transpose_bf16<<<nb((long)Ee*Aa), TPB, 0, stream>>>(psi, psiT_bf, Ee, Aa);

  // ---- one-time: comp_enc = tanh(listener @ psi + b_psi), stored bf16 ----
  wmma_gemm_bf16<Ee><<<gemm_blocks(Bb*Tt, Aa), 256, 0, stream>>>(
      lis_bf, Ee, psiT_bf, b_psi,
      nullptr, 0, comp_bf, Aa, Bb*Tt, Aa, Ee, /*tanh=*/1);

  // ---- sequential decode: 150 steps ----
  for (int t = 0; t < Ll; ++t) {
    // 1) concatenated input [x_t | ctx | h] in bf16
    prep_step<<<nb((long)Bb*KCAT), TPB, 0, stream>>>(
        embedding, tokens, t, hbuf, ctxbuf, concatA);

    // 2) gates = concat @ Wcat^T + (b_ih + b_hh)    [64 x 4096]
    wmma_gemm_bf16<KCAT><<<gemm_blocks(Bb, 4*Dd), 256, 0, stream>>>(
        concatA, KCAT, wcat_bf, bsum,
        gates, 4 * Dd, nullptr, 0, Bb, 4 * Dd, KCAT, 0);

    // 3) LSTM cell update -> h, c (+ h into hc bf16)
    lstm_update<<<nb((long)Bb*Dd), TPB, 0, stream>>>(gates, hbuf, cbuf, hc_bf);

    // 4) q = tanh(h @ phi)   [64 x 256], bf16 output
    wmma_gemm_bf16<Dd><<<gemm_blocks(Bb, Aa), 256, 0, stream>>>(
        hc_bf, KHC, phiT_bf, nullptr,
        nullptr, 0, q_bf, Aa, Bb, Aa, Dd, /*tanh=*/1);

    // 5) masked scaled softmax attention weights
    attn_softmax<<<Bb, 256, 0, stream>>>(comp_bf, q_bf, state_len, attn);

    // 6) ctx = attn @ listener (bf16 L2 stream) -> ctx fp32 + hc bf16
    ctx_kernel<<<Bb * 4, 256, 0, stream>>>(attn, lis_bf, ctxbuf, hc_bf);

    // 7) logits[:, t, :] = [h|ctx] @ W_cls + b_cls   [64 x 8000]
    wmma_gemm_bf16<KHC><<<gemm_blocks(Bb, Vv), 256, 0, stream>>>(
        hc_bf, KHC, wcls_bf, b_cls,
        out + (size_t)t * Vv, (long)Ll * Vv, nullptr, 0,
        Bb, Vv, KHC, 0);
  }
}
